// BiLSTM_88330297409631
// MI455X (gfx1250) — compile-verified, matrix operand order checked by reading, not run on hardware
//
#include <hip/hip_runtime.h>
#include <hip/hip_bf16.h>

// ---------------------------------------------------------------------------
// BiLSTM on gfx1250: WMMA bf16 (f32 accum) for both the input-projection GEMM
// and the per-timestep recurrent GEMM; LDS-resident h/c/gates in a persistent
// scan kernel (one workgroup per direction). Hardware f32<->bf16 conversion;
// all hot-loop operands are b128 vector loads in fragment layout.
// ---------------------------------------------------------------------------

#define BB 64
#define TT 512
#define DD 256
#define HH 256
#define G4H 1024   // 4*H

typedef __attribute__((ext_vector_type(16))) __bf16 v16bf;
typedef __attribute__((ext_vector_type(8)))  float  v8f;

union Frag  { v16bf v; uint4 q[2]; unsigned short s[16]; };
union Pack8 { uint4 q; unsigned short s[8]; };

static __device__ __forceinline__ unsigned short f32_to_bf_bits(float f) {
    __bf16 b = (__bf16)f;                       // hardware v_cvt (RNE)
    return __builtin_bit_cast(unsigned short, b);
}
static __device__ __forceinline__ float bf_bits_to_f32(unsigned short u) {
    return __builtin_bit_cast(float, ((unsigned int)u) << 16);
}
static __device__ __forceinline__ float sigmoid_f(float x) {
    return 1.f / (1.f + __expf(-x));
}
static __device__ __forceinline__ float tanh_f(float x) {
    x = fminf(15.f, fmaxf(-15.f, x));
    float e = __expf(2.f * x);
    return (e - 1.f) / (e + 1.f);
}

// ---------------------------------------------------------------------------
// Zero the output (masking beyond valid lengths relies on this).
// ---------------------------------------------------------------------------
__global__ void zero_kernel(float* __restrict__ out, int n) {
    for (int i = blockIdx.x * blockDim.x + threadIdx.x; i < n;
         i += gridDim.x * blockDim.x)
        out[i] = 0.f;
}

// ---------------------------------------------------------------------------
// Convert x (f32 [B*T*D]) to a bf16 copy once; 8 elements per thread.
// ---------------------------------------------------------------------------
__global__ void xcvt_kernel(const float* __restrict__ x,
                            unsigned short* __restrict__ xb, int n8) {
    int i = blockIdx.x * blockDim.x + threadIdx.x;
    if (i >= n8) return;
    const float* p = x + i * 8;
    float4 f0 = *(const float4*)(p);
    float4 f1 = *(const float4*)(p + 4);
    Pack8 o;
    o.s[0] = f32_to_bf_bits(f0.x); o.s[1] = f32_to_bf_bits(f0.y);
    o.s[2] = f32_to_bf_bits(f0.z); o.s[3] = f32_to_bf_bits(f0.w);
    o.s[4] = f32_to_bf_bits(f1.x); o.s[5] = f32_to_bf_bits(f1.y);
    o.s[6] = f32_to_bf_bits(f1.z); o.s[7] = f32_to_bf_bits(f1.w);
    *(uint4*)(xb + i * 8) = o.q;
}

// ---------------------------------------------------------------------------
// Pack Wih/Whh (f32 [4H][K]) into bf16 WMMA B-fragment layout:
//   [dir][ct(64)][kk(8)][lane(32)][e(16)],  B[k][n] = W[n][k]
//   lane: n = ct*16 + (lane&15), khalf = lane>>4
//   e<8 : k = kk*32 + khalf*8 + e ; e>=8: k = kk*32 + 16 + khalf*8 + (e-8)
// Also bias[dir][n] = bih + bhh.
// ---------------------------------------------------------------------------
__global__ void prep_kernel(const float* __restrict__ Wih_f, const float* __restrict__ Whh_f,
                            const float* __restrict__ bih_f, const float* __restrict__ bhh_f,
                            const float* __restrict__ Wih_b, const float* __restrict__ Whh_b,
                            const float* __restrict__ bih_b, const float* __restrict__ bhh_b,
                            unsigned short* __restrict__ wih_pack,
                            unsigned short* __restrict__ whh_pack,
                            float* __restrict__ bias) {
    int p = blockIdx.x * blockDim.x + threadIdx.x;
    if (p < 2 * G4H) {
        int dir = p >> 10, n = p & 1023;
        bias[p] = dir ? (bih_b[n] + bhh_b[n]) : (bih_f[n] + bhh_f[n]);
    }
    if (p >= (1 << 20)) return;
    int e    = p & 15;
    int lane = (p >> 4) & 31;
    int kk   = (p >> 9) & 7;
    int ct   = (p >> 12) & 63;
    int mat  = (p >> 18) & 1;   // 0 = Wih, 1 = Whh
    int dir  = (p >> 19) & 1;
    int n     = ct * 16 + (lane & 15);
    int khalf = lane >> 4;
    int k     = kk * 32 + khalf * 8 + ((e < 8) ? e : (16 + (e - 8)));
    const float* W = mat ? (dir ? Whh_b : Whh_f) : (dir ? Wih_b : Wih_f);
    unsigned short* dst = mat ? whh_pack : wih_pack;
    dst[dir * 262144 + ((ct * 8 + kk) * 32 + lane) * 16 + e] =
        f32_to_bf_bits(W[n * 256 + k]);
}

// ---------------------------------------------------------------------------
// x_proj (bf16) = x[b, src_s, :] @ WihT, stored step-major & batch-contiguous
// to match the WMMA C/D fragment:  xp_t[dir][s][n(1024)][b(64)].
// Backward direction folds the faithful reversed-input gather:
//   src_s = (s < len) ? len-1-s : 511-(s-len)
// One wave computes one 16x16 output tile over K=256 (8 WMMA k-steps); both
// A and B fragments are two global_load_b128 each (x pre-converted to bf16).
// ---------------------------------------------------------------------------
__global__ __launch_bounds__(256)
void xproj_kernel(const unsigned short* __restrict__ xb, const int* __restrict__ lens,
                  const unsigned short* __restrict__ wih_pack,
                  unsigned short* __restrict__ xp) {
    int g    = blockIdx.x * 8 + (threadIdx.x >> 5);
    int lane = threadIdx.x & 31;
    int l16 = lane & 15, khalf = lane >> 4;

    int dir = g / (2048 * 64);
    int rem = g % (2048 * 64);
    int rt = rem / 64, ct = rem % 64;

    // Per-lane A row (B*T index) and source timestep.
    int r = rt * 16 + l16;
    int b = r >> 9, s = r & 511;
    int src_s;
    if (dir == 0) {
        src_s = s;
    } else {
        int len = lens[b];
        src_s = (s < len) ? (len - 1 - s) : (511 - (s - len));
    }
    const unsigned short* xrow = xb + (b * TT + src_s) * DD;
    const unsigned short* wpb  = wih_pack + dir * 262144 + ct * (8 * 32 * 16);

    v8f acc = (v8f)0.f;
#pragma unroll
    for (int kk = 0; kk < 8; ++kk) {
        int k0 = kk * 32 + khalf * 8;
        Frag a, bf;
        a.q[0] = *(const uint4*)(xrow + k0);
        a.q[1] = *(const uint4*)(xrow + k0 + 16);
        const uint4* p = (const uint4*)(wpb + (kk * 32 + lane) * 16);
        bf.q[0] = p[0]; bf.q[1] = p[1];
        acc = __builtin_amdgcn_wmma_f32_16x16x32_bf16(
            false, a.v, false, bf.v, (short)0, acc, false, false);
    }
    int n = ct * 16 + l16;
#pragma unroll
    for (int j = 0; j < 8; ++j) {
        int rr = rt * 16 + khalf * 8 + j;        // C/D layout: M = j + 8*khalf
        int bj = rr >> 9, sj = rr & 511;         // tile rows = 16 steps of one batch
        xp[((dir * TT + sj) * G4H + n) * 64 + bj] = f32_to_bf_bits(acc[j]);
    }
}

// ---------------------------------------------------------------------------
// Persistent recurrent scan: one workgroup (512 thr = 16 waves) per direction.
// LDS: gates[1024][64] bf16 (128K) + c[64][256] f32 (64K) + h[64][256] bf16
// (32K) + len[64] = 224.25 KB of the WGP's 320 KB.
// Per step: 16 waves x 16 tiles x 8 k-steps of v_wmma_f32_16x16x32_bf16,
// x_proj addend read as 16 b128 loads (fragment-layout xp_t), then an
// elementwise phase combining i,f,g,o -> c,h, scattering h to d_out
// (backward scatters to out[b, len-1-s, 256:]; positions >= len stay zero).
// ---------------------------------------------------------------------------
__global__ __launch_bounds__(512, 1)
void bilstm_scan_kernel(const unsigned short* __restrict__ whh_pack,
                        const float* __restrict__ bias,
                        const unsigned short* __restrict__ xp,
                        const int* __restrict__ lens,
                        float* __restrict__ out) {
    extern __shared__ char smem[];
    unsigned short* gates = (unsigned short*)smem;                       // [1024][64]
    float*          c_lds = (float*)(smem + 131072);                     // [64][256]
    unsigned short* h_lds = (unsigned short*)(smem + 131072 + 65536);    // [64][256]
    int*            len_lds = (int*)(smem + 131072 + 65536 + 32768);     // [64]

    const int dir  = blockIdx.x;
    const int tid  = threadIdx.x;
    const int wv   = tid >> 5;
    const int lane = tid & 31;
    const int l16 = lane & 15, khalf = lane >> 4;

    for (int e = tid; e < BB * HH; e += 512) { c_lds[e] = 0.f; h_lds[e] = 0; }
    if (tid < BB) len_lds[tid] = lens[tid];
    __syncthreads();

    const unsigned short* wp = whh_pack + dir * 262144;
    const bool is_g = ((wv >> 2) == 2);   // waves 8..11 own the g-gate columns

    // Loop-invariant per-lane gate columns and biases.
    int   ncol[4];
    float bv[4];
#pragma unroll
    for (int ci = 0; ci < 4; ++ci) {
        ncol[ci] = (wv * 4 + ci) * 16 + l16;
        bv[ci]   = bias[dir * G4H + ncol[ci]];
    }

    v8f acc[4][4];
    for (int s = 0; s < TT; ++s) {
        // ---- phase 1: gates = act(h @ WhhT + x_proj + bias) --------------
#pragma unroll
        for (int rt = 0; rt < 4; ++rt)
#pragma unroll
            for (int ci = 0; ci < 4; ++ci) acc[rt][ci] = (v8f)0.f;

#pragma unroll
        for (int kk = 0; kk < 8; ++kk) {
            int k0 = kk * 32 + khalf * 8;
            Frag a[4], bf[4];
#pragma unroll
            for (int rt = 0; rt < 4; ++rt) {
                int row = rt * 16 + l16;
                a[rt].q[0] = *(const uint4*)(h_lds + row * HH + k0);
                a[rt].q[1] = *(const uint4*)(h_lds + row * HH + k0 + 16);
            }
#pragma unroll
            for (int ci = 0; ci < 4; ++ci) {
                int ct = wv * 4 + ci;
                const uint4* p = (const uint4*)(wp + ((ct * 8 + kk) * 32 + lane) * 16);
                bf[ci].q[0] = p[0]; bf[ci].q[1] = p[1];
            }
#pragma unroll
            for (int rt = 0; rt < 4; ++rt)
#pragma unroll
                for (int ci = 0; ci < 4; ++ci)
                    acc[rt][ci] = __builtin_amdgcn_wmma_f32_16x16x32_bf16(
                        false, a[rt].v, false, bf[ci].v, (short)0,
                        acc[rt][ci], false, false);
        }

        const unsigned short* xpb = xp + (size_t)((dir * TT + s) * G4H) * 64;
#pragma unroll
        for (int ci = 0; ci < 4; ++ci) {
            int n = ncol[ci];
#pragma unroll
            for (int rt = 0; rt < 4; ++rt) {
                Pack8 xq, op;
                xq.q = *(const uint4*)(xpb + n * 64 + rt * 16 + khalf * 8);
#pragma unroll
                for (int j = 0; j < 8; ++j) {
                    float gpre = acc[rt][ci][j] + bv[ci] + bf_bits_to_f32(xq.s[j]);
                    float ga   = is_g ? tanh_f(gpre) : sigmoid_f(gpre);
                    op.s[j] = f32_to_bf_bits(ga);
                }
                *(uint4*)(gates + n * 64 + rt * 16 + khalf * 8) = op.q;
            }
        }
        __syncthreads();

        // ---- phase 2: c = f*c + i*g ; h = o*tanh(c); scatter to out ------
        for (int e = tid; e < BB * HH; e += 512) {
            int b = e >> 8, j = e & 255;
            float gi = bf_bits_to_f32(gates[(0 * HH + j) * 64 + b]);
            float gf = bf_bits_to_f32(gates[(1 * HH + j) * 64 + b]);
            float gg = bf_bits_to_f32(gates[(2 * HH + j) * 64 + b]);
            float go = bf_bits_to_f32(gates[(3 * HH + j) * 64 + b]);
            float c  = gf * c_lds[e] + gi * gg;
            c_lds[e] = c;
            float hv = go * tanh_f(c);
            h_lds[e] = f32_to_bf_bits(hv);
            int len = len_lds[b];
            if (s < len) {
                int t = dir ? (len - 1 - s) : s;
                out[(b * TT + t) * (2 * HH) + dir * HH + j] = hv;
            }
        }
        __syncthreads();
    }
}

// ---------------------------------------------------------------------------
extern "C" void kernel_launch(void* const* d_in, const int* in_sizes, int n_in,
                              void* d_out, int out_size, void* d_ws, size_t ws_size,
                              hipStream_t stream) {
    const float* x     = (const float*)d_in[0];
    const int*   lens  = (const int*)d_in[1];
    const float* Wih_f = (const float*)d_in[2];
    const float* Whh_f = (const float*)d_in[3];
    const float* bih_f = (const float*)d_in[4];
    const float* bhh_f = (const float*)d_in[5];
    const float* Wih_b = (const float*)d_in[6];
    const float* Whh_b = (const float*)d_in[7];
    const float* bih_b = (const float*)d_in[8];
    const float* bhh_b = (const float*)d_in[9];
    float* out = (float*)d_out;

    char* ws = (char*)d_ws;
    unsigned short* wih_pack = (unsigned short*)ws;                        // 1 MB
    unsigned short* whh_pack = (unsigned short*)(ws + (1 << 20));          // 1 MB
    float*          bias     = (float*)(ws + (2 << 20));                   // 8 KB
    unsigned short* xbf      = (unsigned short*)(ws + (2 << 20) + 8192);   // 16 MB
    unsigned short* xp       = (unsigned short*)(ws + (2 << 20) + 8192 + (16 << 20)); // 128 MB

    zero_kernel<<<4096, 256, 0, stream>>>(out, out_size);
    xcvt_kernel<<<4096, 256, 0, stream>>>(x, xbf, (BB * TT * DD) / 8);
    prep_kernel<<<4096, 256, 0, stream>>>(Wih_f, Whh_f, bih_f, bhh_f,
                                          Wih_b, Whh_b, bih_b, bhh_b,
                                          wih_pack, whh_pack, bias);
    xproj_kernel<<<32768, 256, 0, stream>>>(xbf, lens, wih_pack, xp);

    size_t smem = 131072 + 65536 + 32768 + 256;   // 224.25 KB (<320 KB/WGP)
    bilstm_scan_kernel<<<2, 512, smem, stream>>>(whh_pack, bias, xp, lens, out);
}
